// SingleStreamBlock_1022202216582
// MI455X (gfx1250) — compile-verified, matrix operand order checked by reading, not run on hardware
//
#include <hip/hip_runtime.h>
#include <hip/hip_bf16.h>
#include <math.h>

// Problem dims (SingleStreamBlock): HID=3072, HEADS=24, DH=128, MLP=12288, B=1, L=2048
#define L_SEQ   2048
#define HID     3072
#define HEADS   24
#define DH      128
#define MLPD    12288
#define QKVD    (3 * HID)          // 9216
#define W1R     (3 * HID + MLPD)   // 21504
#define CATD    (HID + MLPD)       // 15360

typedef __bf16 bf16_t;
typedef bf16_t v16bf __attribute__((ext_vector_type(16)));
typedef float  v8f   __attribute__((ext_vector_type(8)));

union FragBF { v16bf v; unsigned short u[16]; };

__device__ __forceinline__ unsigned short f2bf(float f) {
  unsigned int u = __builtin_bit_cast(unsigned int, f);
  u += 0x7fffu + ((u >> 16) & 1u);      // round-to-nearest-even
  return (unsigned short)(u >> 16);
}

// Pack two f32 -> packed bf16x2 (v_cvt_pk_bf16_f32 when available)
__device__ __forceinline__ unsigned int pack_bf16(float a, float b) {
#if __has_builtin(__builtin_amdgcn_cvt_pk_bf16_f32)
  auto r = __builtin_amdgcn_cvt_pk_bf16_f32(a, b);
  return __builtin_bit_cast(unsigned int, r);
#else
  return (unsigned)f2bf(a) | ((unsigned)f2bf(b) << 16);
#endif
}

// ---- CDNA5 async memory->LDS (ASYNCcnt-tracked, bypasses VGPRs) ----
__device__ __forceinline__ void async_load_b128(unsigned int lds_byte_off,
                                                const void* gaddr) {
  asm volatile("global_load_async_to_lds_b128 %0, %1, off"
               :: "v"(lds_byte_off), "v"(gaddr) : "memory");
}
__device__ __forceinline__ void wait_async0() {
#if __has_builtin(__builtin_amdgcn_s_wait_asynccnt)
  __builtin_amdgcn_s_wait_asynccnt(0);
#else
  asm volatile("s_wait_asynccnt 0" ::: "memory");
#endif
}
__device__ __forceinline__ unsigned int lds_off_of(const void* p) {
  return (unsigned int)(unsigned long long)p;   // addr[31:0] = LDS offset
}

// ---- DPP row_ror butterfly all-reduce within each 16-lane row (VALU only) ----
template <int CTRL>
__device__ __forceinline__ float dpp_ror(float v) {
  int i = __builtin_bit_cast(int, v);
  i = __builtin_amdgcn_update_dpp(0, i, CTRL, 0xF, 0xF, true);
  return __builtin_bit_cast(float, i);
}
__device__ __forceinline__ float row16_max(float v) {
  v = fmaxf(v, dpp_ror<0x121>(v));   // row_ror:1
  v = fmaxf(v, dpp_ror<0x122>(v));   // row_ror:2
  v = fmaxf(v, dpp_ror<0x124>(v));   // row_ror:4
  v = fmaxf(v, dpp_ror<0x128>(v));   // row_ror:8
  return v;
}
__device__ __forceinline__ float row16_sum(float v) {
  v += dpp_ror<0x121>(v);
  v += dpp_ror<0x122>(v);
  v += dpp_ror<0x124>(v);
  v += dpp_ror<0x128>(v);
  return v;
}

// K position of pair p for this lane-half (16-bit A-matrix 16x32 layout, ISA 7.12.2)
__device__ __forceinline__ int frag_k(int p, int half) {
  return (p < 4) ? (half * 8 + p * 2) : (16 + half * 8 + (p - 4) * 2);
}

// Gather a 16x32 bf16 A/B fragment from an LDS tile stored [row][k]
__device__ __forceinline__ v16bf frag_row(const unsigned short* lds, int rbase, int ld,
                                          int kbase, int lane) {
  const int r = rbase + (lane & 15);
  const int half = (lane >> 4) & 1;
  FragBF f;
#pragma unroll
  for (int p = 0; p < 8; ++p) {
    const int k = kbase + frag_k(p, half);
    f.u[2 * p]     = lds[r * ld + k];
    f.u[2 * p + 1] = lds[r * ld + k + 1];
  }
  return f.v;
}

// Gather a B fragment from an LDS tile stored [k][col] (V tile in attention)
__device__ __forceinline__ v16bf frag_col(const unsigned short* lds, int cbase, int ld,
                                          int kbase, int lane) {
  const int c = cbase + (lane & 15);
  const int half = (lane >> 4) & 1;
  FragBF f;
#pragma unroll
  for (int p = 0; p < 8; ++p) {
    const int k = kbase + frag_k(p, half);
    f.u[2 * p]     = lds[k * ld + c];
    f.u[2 * p + 1] = lds[(k + 1) * ld + c];
  }
  return f.v;
}

__device__ __forceinline__ v8f wmma_bf16(v16bf a, v16bf b, v8f c) {
  return __builtin_amdgcn_wmma_f32_16x16x32_bf16(false, a, false, b, (short)0, c,
                                                 false, false);
}

// ---------------------------------------------------------------------------
// 1) mod = silu(vec) @ w_mod^T + b_mod   (one wave32 per output row)
// ---------------------------------------------------------------------------
__global__ __launch_bounds__(256) void mod_gemv_kernel(
    const float* __restrict__ vec, const float* __restrict__ w_mod,
    const float* __restrict__ b_mod, float* __restrict__ modout) {
  const int lane = threadIdx.x & 31;
  const int row = blockIdx.x * 8 + (threadIdx.x >> 5);
  float acc = 0.f;
  for (int k = lane; k < HID; k += 32) {
    float v = vec[k];
    float sv = v / (1.f + __expf(-v));
    acc += w_mod[(size_t)row * HID + k] * sv;
  }
#pragma unroll
  for (int m = 16; m >= 1; m >>= 1) acc += __shfl_xor(acc, m, 32);
  if (lane == 0) modout[row] = acc + b_mod[row];
}

// ---------------------------------------------------------------------------
// 2) LayerNorm + (1+scale)*xn + shift -> bf16   (one block per token)
// ---------------------------------------------------------------------------
__global__ __launch_bounds__(256) void ln_mod_kernel(
    const float* __restrict__ x, const float* __restrict__ mod,
    unsigned short* __restrict__ xmod) {
  __shared__ float s1[256], s2[256];
  const int t = blockIdx.x;
  const int tid = threadIdx.x;
  float sum = 0.f, sq = 0.f;
  float vals[HID / 256];
#pragma unroll
  for (int i = 0; i < HID / 256; ++i) {
    float v = x[(size_t)t * HID + tid + i * 256];
    vals[i] = v;
    sum += v;
    sq += v * v;
  }
  s1[tid] = sum; s2[tid] = sq;
  __syncthreads();
  for (int s = 128; s > 0; s >>= 1) {
    if (tid < s) { s1[tid] += s1[tid + s]; s2[tid] += s2[tid + s]; }
    __syncthreads();
  }
  const float mu = s1[0] * (1.f / HID);
  const float var = s2[0] * (1.f / HID) - mu * mu;
  const float rinv = rsqrtf(var + 1e-6f);
#pragma unroll
  for (int i = 0; i < HID / 256; i += 2) {   // pack pairs of adjacent columns
    const int c = tid * 2 + i * 256;         // keep per-thread cols adjacent
    float xn0 = (vals[i] - mu) * rinv;       // vals[i]   maps to col tid + i*256
    float xn1 = (vals[i + 1] - mu) * rinv;
    (void)xn0; (void)xn1; (void)c;
    // fallthrough below with original indexing (simpler, converts each element via f2bf)
  }
#pragma unroll
  for (int i = 0; i < HID / 256; ++i) {
    const int c = tid + i * 256;
    float xn = (vals[i] - mu) * rinv;
    float xm = (1.f + mod[HID + c]) * xn + mod[c];
    xmod[(size_t)t * HID + c] = f2bf(xm);
  }
}

// ---------------------------------------------------------------------------
// 3/6) Generic bf16-WMMA GEMM: C[M,N] = A[M,K](bf16) * W[N,K](f32->bf16) + bias
// ---------------------------------------------------------------------------
template <int MODE>
__global__ __launch_bounds__(256) void gemm_bf16_kernel(
    const unsigned short* __restrict__ A, const float* __restrict__ W,
    const float* __restrict__ bias, int M, int N, int K,
    float* __restrict__ qkv_out, unsigned short* __restrict__ cat_out,
    const float* __restrict__ xres, const float* __restrict__ mod,
    float* __restrict__ out) {
  __shared__ unsigned short As[128 * 32];
  __shared__ unsigned short Bs[64 * 32];
  const int tid = threadIdx.x;
  const int lane = tid & 31;
  const int wave = tid >> 5;
  const int row0 = blockIdx.y * 128;
  const int col0 = blockIdx.x * 64;

  v8f acc[4];
#pragma unroll
  for (int i = 0; i < 4; ++i)
#pragma unroll
    for (int j = 0; j < 8; ++j) acc[i][j] = 0.f;

  const int ar = tid >> 1, ac = (tid & 1) * 16;   // A: 2 thr/row, 16 bf16 each
  const int br = tid >> 2, bc = (tid & 3) * 8;    // B: 4 thr/row, 8 f32 each
  const unsigned int As_off0 = lds_off_of(&As[ar * 32 + ac]);
  const unsigned int As_off1 = lds_off_of(&As[ar * 32 + ac + 8]);

  for (int k0 = 0; k0 < K; k0 += 32) {
    __syncthreads();
    // A tile 128x32 bf16: async memory->LDS (no VGPR round trip)
    const unsigned short* asrc = A + (size_t)(row0 + ar) * K + k0 + ac;
    async_load_b128(As_off0, asrc);
    async_load_b128(As_off1, asrc + 8);
    // B tile 64x32: stream f32 weights, convert to bf16 in-register
    const float4* wsrc = (const float4*)(W + (size_t)(col0 + br) * K + k0 + bc);
    float4 f0 = wsrc[0], f1 = wsrc[1];
    uint4 packed;
    packed.x = pack_bf16(f0.x, f0.y);
    packed.y = pack_bf16(f0.z, f0.w);
    packed.z = pack_bf16(f1.x, f1.y);
    packed.w = pack_bf16(f1.z, f1.w);
    *(uint4*)&Bs[br * 32 + bc] = packed;
    if (k0 + 32 < K)  // prefetch next weight tile (global_prefetch_b8)
      __builtin_prefetch(W + (size_t)(col0 + br) * K + k0 + 32 + bc, 0, 1);
    wait_async0();
    __syncthreads();

    v16bf a = frag_row(As, wave * 16, 32, 0, lane);
#pragma unroll
    for (int nt = 0; nt < 4; ++nt) {
      v16bf b = frag_row(Bs, nt * 16, 32, 0, lane);
      acc[nt] = wmma_bf16(a, b, acc[nt]);
    }
  }

  const int half = (lane >> 4) & 1;
#pragma unroll
  for (int nt = 0; nt < 4; ++nt) {
    const int col = col0 + nt * 16 + (lane & 15);
#pragma unroll
    for (int j = 0; j < 8; ++j) {
      const int row = row0 + wave * 16 + j + 8 * half;
      float v = acc[nt][j] + bias[col];
      if (MODE == 0) {
        if (col < QKVD) {
          qkv_out[(size_t)row * QKVD + col] = v;
        } else {
          float g = 0.5f * v *
                    (1.f + tanhf(0.7978845608f * (v + 0.044715f * v * v * v)));
          cat_out[(size_t)row * CATD + HID + (col - QKVD)] = f2bf(g);
        }
      } else {
        const float gate = mod[2 * HID + col];
        out[(size_t)row * HID + col] =
            xres[(size_t)row * HID + col] + gate * v;
      }
    }
  }
}

// ---------------------------------------------------------------------------
// 4) RMSNorm(q,k) + RoPE + bf16 pack; v -> bf16. One wave per (token, head).
//    1/sqrt(DH) softmax scale is folded into q here.
// ---------------------------------------------------------------------------
__global__ __launch_bounds__(256) void rmsrope_kernel(
    const float* __restrict__ qkv, const float* __restrict__ pe,
    const float* __restrict__ q_scale, const float* __restrict__ k_scale,
    unsigned short* __restrict__ qb, unsigned short* __restrict__ kb,
    unsigned short* __restrict__ vb) {
  const int lane = threadIdx.x & 31;
  const int gid = blockIdx.x * 8 + (threadIdx.x >> 5);
  const int t = gid / HEADS;
  const int h = gid % HEADS;
  const int d0 = lane * 4;
  const size_t dstbase = ((size_t)h * L_SEQ + t) * DH + d0;

  const float4 pe0 = *(const float4*)(pe + ((size_t)t * 64 + (d0 >> 1)) * 4);
  const float4 pe1 = *(const float4*)(pe + ((size_t)t * 64 + (d0 >> 1) + 1) * 4);

#pragma unroll
  for (int s = 0; s < 2; ++s) {  // 0: q (pre-scaled), 1: k
    const float* scl = (s == 0) ? q_scale : k_scale;
    const float post = (s == 0) ? 0.088388347648318447f : 1.f;  // 1/sqrt(128)
    float4 v = *(const float4*)(qkv + (size_t)t * QKVD + s * HID + h * DH + d0);
    float ss = v.x * v.x + v.y * v.y + v.z * v.z + v.w * v.w;
#pragma unroll
    for (int m = 16; m >= 1; m >>= 1) ss += __shfl_xor(ss, m, 32);
    const float r = rsqrtf(ss * (1.f / DH) + 1e-6f) * post;
    float x0 = v.x * r * scl[d0 + 0];
    float x1 = v.y * r * scl[d0 + 1];
    float x2 = v.z * r * scl[d0 + 2];
    float x3 = v.w * r * scl[d0 + 3];
    float o0 = pe0.x * x0 + pe0.y * x1;
    float o1 = pe0.z * x0 + pe0.w * x1;
    float o2 = pe1.x * x2 + pe1.y * x3;
    float o3 = pe1.z * x2 + pe1.w * x3;
    unsigned short* dst = (s == 0) ? qb : kb;
    uint2 p;
    p.x = pack_bf16(o0, o1);
    p.y = pack_bf16(o2, o3);
    *(uint2*)&dst[dstbase] = p;
  }
  float4 v = *(const float4*)(qkv + (size_t)t * QKVD + 2 * HID + h * DH + d0);
  uint2 p;
  p.x = pack_bf16(v.x, v.y);
  p.y = pack_bf16(v.z, v.w);
  *(uint2*)&vb[dstbase] = p;
}

// ---------------------------------------------------------------------------
// 5) Flash attention with WMMA. Async LDS staging, DPP softmax reductions.
// ---------------------------------------------------------------------------
__global__ __launch_bounds__(256) void attn_kernel(
    const unsigned short* __restrict__ Qb, const unsigned short* __restrict__ Kb,
    const unsigned short* __restrict__ Vb, unsigned short* __restrict__ cat_out) {
  __shared__ unsigned short Qs[128 * DH];   // 32 KB
  __shared__ unsigned short Ks[32 * DH];    // 8 KB
  __shared__ unsigned short Vs[32 * DH];    // 8 KB
  __shared__ unsigned short Ps[8 * 16 * 32];// 8 KB (per-wave P scratch)
  const int tid = threadIdx.x;
  const int lane = tid & 31;
  const int wave = tid >> 5;
  const int half = (lane >> 4) & 1;
  const int head = blockIdx.y;
  const int q0 = blockIdx.x * 128;
  const unsigned short* Qh = Qb + (size_t)head * L_SEQ * DH;
  const unsigned short* Kh = Kb + (size_t)head * L_SEQ * DH;
  const unsigned short* Vh = Vb + (size_t)head * L_SEQ * DH;

  // stage Q tile (128 x 128) via async copy
#pragma unroll
  for (int rpt = 0; rpt < 8; ++rpt) {
    int id = rpt * 256 + tid;
    int r = id >> 4, c = (id & 15) * 8;
    async_load_b128(lds_off_of(&Qs[r * DH + c]), Qh + (size_t)(q0 + r) * DH + c);
  }
  wait_async0();
  __syncthreads();
  v16bf qf[4];
#pragma unroll
  for (int kk = 0; kk < 4; ++kk) qf[kk] = frag_row(Qs, wave * 16, DH, kk * 32, lane);

  float run_max[8], run_sum[8];
  v8f o[8];
#pragma unroll
  for (int j = 0; j < 8; ++j) { run_max[j] = -3.0e38f; run_sum[j] = 0.f; }
#pragma unroll
  for (int nt = 0; nt < 8; ++nt)
#pragma unroll
    for (int j = 0; j < 8; ++j) o[nt][j] = 0.f;

  unsigned short* Pw = &Ps[wave * 16 * 32];
  const int kvr = tid >> 4, kvc = (tid & 15) * 8;   // K/V staging coords (2x/thread)

  for (int kt = 0; kt < L_SEQ / 32; ++kt) {
    __syncthreads();
#pragma unroll
    for (int rpt = 0; rpt < 2; ++rpt) {
      const int r = rpt * 16 + kvr;
      async_load_b128(lds_off_of(&Ks[r * DH + kvc]),
                      Kh + (size_t)(kt * 32 + r) * DH + kvc);
      async_load_b128(lds_off_of(&Vs[r * DH + kvc]),
                      Vh + (size_t)(kt * 32 + r) * DH + kvc);
    }
    wait_async0();
    __syncthreads();

    v8f s0, s1;
#pragma unroll
    for (int j = 0; j < 8; ++j) { s0[j] = 0.f; s1[j] = 0.f; }
#pragma unroll
    for (int kk = 0; kk < 4; ++kk) {
      s0 = wmma_bf16(qf[kk], frag_row(Ks, 0, DH, kk * 32, lane), s0);
      s1 = wmma_bf16(qf[kk], frag_row(Ks, 16, DH, kk * 32, lane), s1);
    }

#pragma unroll
    for (int j = 0; j < 8; ++j) {
      const float a0 = s0[j], a1 = s1[j];          // q pre-scaled by 1/sqrt(DH)
      const float m = row16_max(fmaxf(a0, a1));    // DPP, no LDS traffic
      const float mn = fmaxf(run_max[j], m);
      const float alpha = __expf(run_max[j] - mn);
      const float p0 = __expf(a0 - mn), p1 = __expf(a1 - mn);
      const float ps = row16_sum(p0 + p1);
      run_sum[j] = run_sum[j] * alpha + ps;
      run_max[j] = mn;
#pragma unroll
      for (int nt = 0; nt < 8; ++nt) o[nt][j] *= alpha;
      const int prow = j + 8 * half;
      Pw[prow * 32 + (lane & 15)] = f2bf(p0);
      Pw[prow * 32 + 16 + (lane & 15)] = f2bf(p1);
    }
    __syncthreads();  // make P stores visible before A-layout gather
    v16bf pf = frag_row(Pw, 0, 32, 0, lane);
#pragma unroll
    for (int nt = 0; nt < 8; ++nt)
      o[nt] = wmma_bf16(pf, frag_col(Vs, nt * 16, DH, 0, lane), o[nt]);
  }

#pragma unroll
  for (int j = 0; j < 8; ++j) {
    const int qrow = q0 + wave * 16 + j + 8 * half;
    const float rs = 1.f / run_sum[j];
#pragma unroll
    for (int nt = 0; nt < 8; ++nt) {
      cat_out[(size_t)qrow * CATD + head * DH + nt * 16 + (lane & 15)] =
          f2bf(o[nt][j] * rs);
    }
  }
}

// ---------------------------------------------------------------------------
extern "C" void kernel_launch(void* const* d_in, const int* in_sizes, int n_in,
                              void* d_out, int out_size, void* d_ws, size_t ws_size,
                              hipStream_t stream) {
  const float* x       = (const float*)d_in[0];
  const float* vec     = (const float*)d_in[1];
  const float* pe      = (const float*)d_in[2];
  const float* w1      = (const float*)d_in[3];
  const float* b1      = (const float*)d_in[4];
  const float* w2      = (const float*)d_in[5];
  const float* b2      = (const float*)d_in[6];
  const float* w_mod   = (const float*)d_in[7];
  const float* b_mod   = (const float*)d_in[8];
  const float* q_scale = (const float*)d_in[9];
  const float* k_scale = (const float*)d_in[10];
  float* out = (float*)d_out;
  (void)in_sizes; (void)n_in; (void)out_size; (void)ws_size;

  char* ws = (char*)d_ws;
  auto align256 = [](size_t v) { return (v + 255) & ~(size_t)255; };
  size_t off = 0;
  float* modw = (float*)(ws + off);            off += align256((size_t)QKVD * 4);
  unsigned short* xmod = (unsigned short*)(ws + off); off += align256((size_t)L_SEQ * HID * 2);
  float* qkv = (float*)(ws + off);             off += align256((size_t)L_SEQ * QKVD * 4);
  unsigned short* cat = (unsigned short*)(ws + off);  off += align256((size_t)L_SEQ * CATD * 2);
  unsigned short* qb = (unsigned short*)(ws + off);   off += align256((size_t)HEADS * L_SEQ * DH * 2);
  unsigned short* kb = (unsigned short*)(ws + off);   off += align256((size_t)HEADS * L_SEQ * DH * 2);
  unsigned short* vb = (unsigned short*)(ws + off);   off += align256((size_t)HEADS * L_SEQ * DH * 2);

  mod_gemv_kernel<<<QKVD / 8, 256, 0, stream>>>(vec, w_mod, b_mod, modw);
  ln_mod_kernel<<<L_SEQ, 256, 0, stream>>>(x, modw, xmod);
  gemm_bf16_kernel<0><<<dim3(W1R / 64, L_SEQ / 128), 256, 0, stream>>>(
      xmod, w1, b1, L_SEQ, W1R, HID, qkv, cat, nullptr, nullptr, nullptr);
  rmsrope_kernel<<<L_SEQ * HEADS / 8, 256, 0, stream>>>(qkv, pe, q_scale, k_scale,
                                                        qb, kb, vb);
  attn_kernel<<<dim3(L_SEQ / 128, HEADS), 256, 0, stream>>>(qb, kb, vb, cat);
  gemm_bf16_kernel<1><<<dim3(HID / 64, L_SEQ / 128), 256, 0, stream>>>(
      cat, w2, b2, L_SEQ, HID, CATD, nullptr, nullptr, x, modw, out);
}
